// Attention_90787018703279
// MI455X (gfx1250) — matrix operand order repaired and checked by reading, not compiled, not run on hardware
//
#include <hip/hip_runtime.h>
#include <hip/hip_bf16.h>
#include <math.h>

// Problem sizes (fixed by the reference)
static constexpr int Bn = 64;
static constexpr int Sn = 2048;
static constexpr int Dn = 512;
static constexpr int Hn = 512;

typedef __attribute__((ext_vector_type(16))) __bf16 v16bf;
typedef __attribute__((ext_vector_type(8)))  float  v8f;

// Hardware transcendental tanh (gfx1250 V_TANH_F32) with portable fallback.
__device__ __forceinline__ float fast_tanh(float x) {
#if __has_builtin(__builtin_amdgcn_tanhf)
    return __builtin_amdgcn_tanhf(x);
#else
    return tanhf(x);
#endif
}

// ---------------------------------------------------------------------------
// Kernel 1: convert We (f32, [H,D] row-major) to bf16 once; reused by all blocks.
// ---------------------------------------------------------------------------
__global__ __launch_bounds__(256) void conv_bf16_kernel(const float* __restrict__ src,
                                                        __bf16* __restrict__ dst, int n) {
    int i = blockIdx.x * 256 + threadIdx.x;
    if (i < n) dst[i] = (__bf16)src[i];
}

// ---------------------------------------------------------------------------
// Kernel 2: di = hidden @ Wd^T + bd   [B,H]  (tiny: 33 MFLOP)
// ---------------------------------------------------------------------------
__global__ __launch_bounds__(256) void di_kernel(const float* __restrict__ hidden,
                                                 const float* __restrict__ Wd,
                                                 const float* __restrict__ bd,
                                                 float* __restrict__ di) {
    int idx = blockIdx.x * 256 + threadIdx.x;     // b*H + h
    int b = idx >> 9;
    int h = idx & (Hn - 1);
    const float4* hp = (const float4*)(hidden + (size_t)b * Dn);
    const float4* wp = (const float4*)(Wd + (size_t)h * Dn);
    float acc = bd[h];
#pragma unroll 4
    for (int i = 0; i < Dn / 4; ++i) {
        float4 hv = hp[i], wv = wp[i];
        acc += hv.x * wv.x + hv.y * wv.y + hv.z * wv.z + hv.w * wv.w;
    }
    di[idx] = acc;
}

// ---------------------------------------------------------------------------
// Kernel 3 (MAIN, WMMA): per block -> 16 rows of S, full H=512.
//   ei_tile = ctx_tile @ We^T + be  (bf16 WMMA, f32 accum, never stored)
//   u[s]    = sum_h V[h]*tanh(di[b,h] + ei[s,h])
// 8 waves/block, each wave owns 4 n-tiles (64 columns). K loop = 16 WMMAs/tile.
// A fragment (16x32 bf16): lane (0-15) row M=lane, K=kt+0..15;
//                          lane (16-31) row M=lane-16, K=kt+16..31.
// B fragment (32x16 bf16): lane holds column n=lane&15, K=kt+(lane>>4)*16+0..15
//                          = 16 contiguous d's of We row n (row-major friendly).
// ---------------------------------------------------------------------------
__global__ __launch_bounds__(256) void ei_score_kernel(const float* __restrict__ ctx,   // [B,S,D]
                                                       const __bf16* __restrict__ WeB,  // [H,D] bf16
                                                       const float* __restrict__ be,    // [H]
                                                       const float* __restrict__ di,    // [B,H]
                                                       const float* __restrict__ V,     // [H]
                                                       float* __restrict__ u) {         // [B,S]
    const int b    = blockIdx.y;
    const int m0   = blockIdx.x * 16;            // s-tile base
    const int wave = threadIdx.x >> 5;           // 0..7
    const int lane = threadIdx.x & 31;
    const int lrow = lane & 15;
    const int lhi  = lane >> 4;
    const int n0   = wave * 64;                  // this wave's column base

    // Per-lane A row pointer: row = m0 + lrow, element offset lhi*16 (contiguous K, matches B)
    const float* arow = ctx + ((size_t)b * Sn + (m0 + lrow)) * Dn + lhi * 16;

    v8f acc[4];
#pragma unroll
    for (int nt = 0; nt < 4; ++nt) {
        float bev = be[n0 + nt * 16 + lrow];     // fold +be into the accumulator
#pragma unroll
        for (int r = 0; r < 8; ++r) acc[nt][r] = bev;
    }

    for (int kt = 0; kt < Dn; kt += 32) {
        // ---- build bf16 A fragment from f32 context (two 32B spans, aligned) ----
        const float4* ap0 = (const float4*)(arow + kt);
        const float4* ap1 = (const float4*)(arow + kt + 8);
        float4 f0 = ap0[0], f1 = ap0[1];
        float4 f2 = ap1[0], f3 = ap1[1];
        v16bf a;
        a[0]  = (__bf16)f0.x; a[1]  = (__bf16)f0.y; a[2]  = (__bf16)f0.z; a[3]  = (__bf16)f0.w;
        a[4]  = (__bf16)f1.x; a[5]  = (__bf16)f1.y; a[6]  = (__bf16)f1.z; a[7]  = (__bf16)f1.w;
        a[8]  = (__bf16)f2.x; a[9]  = (__bf16)f2.y; a[10] = (__bf16)f2.z; a[11] = (__bf16)f2.w;
        a[12] = (__bf16)f3.x; a[13] = (__bf16)f3.y; a[14] = (__bf16)f3.z; a[15] = (__bf16)f3.w;

#pragma unroll
        for (int nt = 0; nt < 4; ++nt) {
            // B fragment: 16 contiguous bf16 from We row (n0+nt*16+lrow), 32B aligned
            const __bf16* bp = WeB + (size_t)(n0 + nt * 16 + lrow) * Dn + kt + lhi * 16;
            v16bf bb = *(const v16bf*)bp;
            acc[nt] = __builtin_amdgcn_wmma_f32_16x16x32_bf16(
                false, a, false, bb, (short)0, acc[nt], false, false);
        }
    }

    // ---- epilogue: u[s] = sum_n V[n]*tanh(di[n] + ei[s,n]); cross-lane/wave via LDS ----
    __shared__ float uls[16];
    if (threadIdx.x < 16) uls[threadIdx.x] = 0.0f;
    __syncthreads();

    const float* dib = di + (size_t)b * Hn;
    float part[8];
#pragma unroll
    for (int r = 0; r < 8; ++r) part[r] = 0.0f;
#pragma unroll
    for (int nt = 0; nt < 4; ++nt) {
        int n = n0 + nt * 16 + lrow;
        float dv = dib[n];
        float vv = V[n];
#pragma unroll
        for (int r = 0; r < 8; ++r)
            part[r] += vv * fast_tanh(dv + acc[nt][r]); // C layout: row = r+8*lhi, col = lrow
    }
#pragma unroll
    for (int r = 0; r < 8; ++r)
        atomicAdd(&uls[r + 8 * lhi], part[r]);
    __syncthreads();

    if (threadIdx.x < 16)
        u[(size_t)b * Sn + m0 + threadIdx.x] = uls[threadIdx.x];
}

// ---------------------------------------------------------------------------
// Kernel 4: masked softmax over S per batch row. mask==true -> -inf (alpha=0).
// ---------------------------------------------------------------------------
__global__ __launch_bounds__(256) void softmax_kernel(const float* __restrict__ u,
                                                      const unsigned char* __restrict__ mask,
                                                      float* __restrict__ alpha) {
    const int b = blockIdx.x;
    const int t = threadIdx.x;
    float v[8];
    float mx = -INFINITY;
#pragma unroll
    for (int i = 0; i < 8; ++i) {
        int s = t + i * 256;
        float x = u[(size_t)b * Sn + s];
        v[i] = mask[(size_t)b * Sn + s] ? -INFINITY : x;
        mx = fmaxf(mx, v[i]);
    }
    __shared__ float red[256];
    red[t] = mx;
    __syncthreads();
    for (int st = 128; st > 0; st >>= 1) {
        if (t < st) red[t] = fmaxf(red[t], red[t + st]);
        __syncthreads();
    }
    mx = red[0];
    __syncthreads();
    float sum = 0.0f;
#pragma unroll
    for (int i = 0; i < 8; ++i) { v[i] = __expf(v[i] - mx); sum += v[i]; }
    red[t] = sum;
    __syncthreads();
    for (int st = 128; st > 0; st >>= 1) {
        if (t < st) red[t] += red[t + st];
        __syncthreads();
    }
    float inv = 1.0f / red[0];
#pragma unroll
    for (int i = 0; i < 8; ++i)
        alpha[(size_t)b * Sn + t + i * 256] = v[i] * inv;
}

// ---------------------------------------------------------------------------
// Kernel 5a: zero cbar.  Kernel 5b: cbar[b,d] = sum_s alpha[b,s]*ctx[b,s,d]
// (second and final 268MB pass over context; coalesced over d, LDS-staged alpha;
//  context loads are non-temporal: zero reuse, avoid thrashing L2)
// ---------------------------------------------------------------------------
__global__ __launch_bounds__(256) void zero_kernel(float* __restrict__ p, int n) {
    int i = blockIdx.x * 256 + threadIdx.x;
    if (i < n) p[i] = 0.0f;
}

__global__ __launch_bounds__(512) void cbar_kernel(const float* __restrict__ ctx,
                                                   const float* __restrict__ alpha,
                                                   float* __restrict__ cbar) {
    const int b  = blockIdx.y;
    const int s0 = blockIdx.x * 128;
    const int d  = threadIdx.x;                 // 0..511
    __shared__ float al[128];
    if (threadIdx.x < 128) al[threadIdx.x] = alpha[(size_t)b * Sn + s0 + threadIdx.x];
    __syncthreads();
    const float* cp = ctx + ((size_t)b * Sn + s0) * Dn + d;
    float acc0 = 0.0f, acc1 = 0.0f;
#pragma unroll 4
    for (int s = 0; s < 128; s += 2) {
        acc0 += al[s]     * __builtin_nontemporal_load(cp + (size_t)s * Dn);
        acc1 += al[s + 1] * __builtin_nontemporal_load(cp + (size_t)(s + 1) * Dn);
    }
    atomicAdd(&cbar[b * Dn + d], acc0 + acc1);
}

// ---------------------------------------------------------------------------
// Kernel 6: attn[b,h] = cbar[b,:] . We[h,:] + be[h]   (exact refactor of
// sum_s alpha*ei since sum_s alpha == 1). Tiny; keep f32 for output precision.
// ---------------------------------------------------------------------------
__global__ __launch_bounds__(256) void attn_out_kernel(const float* __restrict__ cbar,
                                                       const float* __restrict__ We,
                                                       const float* __restrict__ be,
                                                       float* __restrict__ out) {
    int idx = blockIdx.x * 256 + threadIdx.x;   // b*H + h
    int b = idx >> 9;
    int h = idx & (Hn - 1);
    const float4* c = (const float4*)(cbar + (size_t)b * Dn);
    const float4* w = (const float4*)(We + (size_t)h * Dn);
    float acc = be[h];
#pragma unroll 4
    for (int i = 0; i < Dn / 4; ++i) {
        float4 cv = c[i], wv = w[i];
        acc += cv.x * wv.x + cv.y * wv.y + cv.z * wv.z + cv.w * wv.w;
    }
    out[idx] = acc;
}

// ---------------------------------------------------------------------------
extern "C" void kernel_launch(void* const* d_in, const int* in_sizes, int n_in,
                              void* d_out, int out_size, void* d_ws, size_t ws_size,
                              hipStream_t stream) {
    (void)in_sizes; (void)n_in; (void)out_size; (void)ws_size;

    const float*         hidden = (const float*)d_in[0];          // [B,D]
    const float*         ctx    = (const float*)d_in[1];          // [B,S,D]
    const unsigned char* mask   = (const unsigned char*)d_in[2];  // [B,S] bool
    const float*         Wd     = (const float*)d_in[3];          // [H,D]
    const float*         bd     = (const float*)d_in[4];          // [H]
    const float*         We     = (const float*)d_in[5];          // [H,D]
    const float*         be     = (const float*)d_in[6];          // [H]
    const float*         V      = (const float*)d_in[7];          // [H]

    float* alpha_out = (float*)d_out;                             // [B,S]
    float* attn_out  = (float*)d_out + (size_t)Bn * Sn;           // [B,H]

    // Workspace layout (total ~1.3 MB)
    char*   ws   = (char*)d_ws;
    __bf16* WeB  = (__bf16*)ws;                                   // 512 KB
    float*  di   = (float*)(ws + 524288);                         // 128 KB
    float*  u    = (float*)(ws + 524288 + 131072);                // 512 KB
    float*  cbar = (float*)(ws + 524288 + 131072 + 524288);       // 128 KB

    conv_bf16_kernel<<<(Hn * Dn) / 256, 256, 0, stream>>>(We, WeB, Hn * Dn);
    di_kernel<<<(Bn * Hn) / 256, 256, 0, stream>>>(hidden, Wd, bd, di);
    zero_kernel<<<(Bn * Dn) / 256, 256, 0, stream>>>(cbar, Bn * Dn);

    ei_score_kernel<<<dim3(Sn / 16, Bn), 256, 0, stream>>>(ctx, WeB, be, di, V, u);
    softmax_kernel<<<Bn, 256, 0, stream>>>(u, mask, alpha_out);
    cbar_kernel<<<dim3(Sn / 128, Bn), 512, 0, stream>>>(ctx, alpha_out, cbar);
    attn_out_kernel<<<(Bn * Hn) / 256, 256, 0, stream>>>(cbar, We, be, attn_out);
}
